// MFLayer_38757784879229
// MI455X (gfx1250) — matrix operand order repaired and checked by reading, not compiled, not run on hardware
//
#include <hip/hip_runtime.h>
#include <hip/hip_bf16.h>

// ---------------------------------------------------------------------------
// MFLayer for MI455X (gfx1250, wave32, WMMA).
//
// Math: both conv1d(pad=len-1)+mod_fold stages are circular correlations.
//   Stage1: d[m,j]      = sum_k S[k] * x[m,(j+k+1) mod D]        (m = b*2+c)
//   Stage2: dh[b,t,c,j] = sum_k H[t,c,k] * g[m, j+k],
//           g[m,p] = d[m,(p-K+1) mod D],  p in [0, D+K-1)   (Hankel GEMM)
// Stage2 is 16 GEMMs of 4096x32768x4096 = 17.6 TFLOP; output is 8.6 GB f32
// => memory-bound (~0.37 ms at 23.3 TB/s HBM). Compute on the bf16 WMMA path
// is microseconds, so we spend 3x WMMAs on a 2-term bf16 split (hi+lo) for
// ~fp32 accuracy. Template (64 MB per channel) is kept L2-resident: blockIdx.z
// groups same-channel blocks and the output is written with non-temporal
// stores so the 8.6 GB stream does not evict it.
// ---------------------------------------------------------------------------

typedef __bf16 v16bf  __attribute__((ext_vector_type(16)));
typedef __bf16 bf16x8 __attribute__((ext_vector_type(8)));
typedef float  v8f    __attribute__((ext_vector_type(8)));

#define D_LEN   32768
#define K_LEN   4096
#define T_LEN   4096
#define B_LEN   8
#define G_LEN   36864   // D + K - 1 = 36863, rounded up

// ---------------------------------------------------------------------------
// Stage 1: circular correlation -> g (bf16 hi/lo), wrap-extended Hankel vector
// grid (D/256, 16), block 256
// ---------------------------------------------------------------------------
__global__ __launch_bounds__(256) void mf_stage1(const float* __restrict__ X,
                                                 const float* __restrict__ S,
                                                 __bf16* __restrict__ gHi,
                                                 __bf16* __restrict__ gLo) {
    const int m   = blockIdx.y;          // b*2 + c (matches X layout)
    const int j0  = blockIdx.x * 256;
    const int tid = threadIdx.x;

    __shared__ float sS[256];
    __shared__ float sX[512];

    const float* x = X + (size_t)m * D_LEN;
    float acc = 0.0f;

    for (int kb = 0; kb < D_LEN; kb += 256) {
        __syncthreads();
        sS[tid] = S[kb + tid];
        const int base = j0 + kb + 1 + tid;
        sX[tid]       = x[ base        & (D_LEN - 1)];
        sX[tid + 256] = x[(base + 256) & (D_LEN - 1)];
        __syncthreads();
#pragma unroll 8
        for (int kk = 0; kk < 256; ++kk)
            acc = fmaf(sS[kk], sX[tid + kk], acc);
    }

    // d[m,j] = acc; scatter into g[p], p == (j + K - 1) mod D (+D copy if in range)
    const int j = j0 + tid;
    const __bf16 hi = (__bf16)acc;
    const __bf16 lo = (__bf16)(acc - (float)hi);
    const size_t gb = (size_t)m * G_LEN;
    const int p0 = (j + (K_LEN - 1)) & (D_LEN - 1);
    gHi[gb + p0] = hi;
    gLo[gb + p0] = lo;
    if (p0 < K_LEN) {                 // p0 + D <= D + K - 1
        gHi[gb + p0 + D_LEN] = hi;
        gLo[gb + p0 + D_LEN] = lo;
    }
}

// ---------------------------------------------------------------------------
// Stage 2: Hankel GEMM with WMMA bf16 split-precision, double-buffered LDS.
// grid (N/128=256, M/128=32, 16), block 256 (8 waves).
// Block tile 128(t) x 128(j); wave tile 32x64 = 2x4 16x16 WMMA tiles; KB=32.
// Pipeline per k-step: issue next A-tile global loads -> WMMA on current
// buffer -> convert/pack next buffer -> single barrier.
// ---------------------------------------------------------------------------
__global__ __launch_bounds__(256) void mf_stage2(const float* __restrict__ tmpl,
                                                 const float* __restrict__ hh,
                                                 const __bf16* __restrict__ gHi,
                                                 const __bf16* __restrict__ gLo,
                                                 float* __restrict__ out) {
    const int j0  = blockIdx.x * 128;
    const int t0  = blockIdx.y * 128;
    const int z   = blockIdx.z;           // c*8 + b  (groups same-c blocks for L2)
    const int c   = z >> 3;
    const int b   = z & 7;
    const int gm  = b * 2 + c;            // g layout index

    const int tid      = threadIdx.x;
    const int lane     = tid & 31;
    const int wave     = tid >> 5;
    const int laneN    = lane & 15;       // N column / M row within fragment
    const int laneHalf = lane >> 4;       // K-half selector
    const int wm       = wave & 3;        // 4 m-slots of 32 rows
    const int wn       = wave >> 2;       // 2 n-slots of 64 cols

    __shared__ __bf16 sGHi[K_LEN + 128];  // g[j0 .. j0+K+127]
    __shared__ __bf16 sGLo[K_LEN + 128];
    __shared__ __bf16 sAHi[2][128 * 32];  // A tile  [row][k], double-buffered
    __shared__ __bf16 sALo[2][128 * 32];
    __shared__ __bf16 sBHi[2][128 * 32];  // B tile transposed: [n][k]
    __shared__ __bf16 sBLo[2][128 * 32];
    __shared__ float  sRcp[128];

    // One-time: g window + reciprocal hh_sqrt for this row block.
    {
        const __bf16* gh = gHi + (size_t)gm * G_LEN + j0;
        const __bf16* gl = gLo + (size_t)gm * G_LEN + j0;
        for (int q = tid; q < K_LEN + 128; q += 256) {
            sGHi[q] = gh[q];
            sGLo[q] = gl[q];
        }
        if (tid < 128)
            sRcp[tid] = 1.0f / hh[(size_t)(t0 + tid) * 2 + c];
    }
    __syncthreads();

    // Row-block base into template: tmpl[(t0+r)*2 + c][k]
    const size_t tBase   = ((size_t)t0 * 2 + c) * K_LEN;
    const size_t rStride = (size_t)2 * K_LEN;

    // Prologue: stage k-tile 0 into buffer 0.
    {
#pragma unroll
        for (int i = 0; i < 16; ++i) {
            const int e = tid + i * 256;       // e = r*32 + k
            const int r = e >> 5;
            const int k = e & 31;
            const float f = tmpl[tBase + (size_t)r * rStride + k];
            const __bf16 ahi = (__bf16)f;
            sAHi[0][e] = ahi;
            sALo[0][e] = (__bf16)(f - (float)ahi);
            sBHi[0][e] = sGHi[r + k];          // B[k][n]=g[k0+k+n], transposed [n=r][k]
            sBLo[0][e] = sGLo[r + k];
        }
    }
    __syncthreads();

    const v8f vzero = {0.f, 0.f, 0.f, 0.f, 0.f, 0.f, 0.f, 0.f};
    v8f acc[2][4];
#pragma unroll
    for (int mi = 0; mi < 2; ++mi)
#pragma unroll
        for (int ni = 0; ni < 4; ++ni)
            acc[mi][ni] = vzero;

    union Frag { v16bf v; bf16x8 h[2]; };

    for (int it = 0; it < K_LEN / 32; ++it) {
        const int  k0   = it * 32;
        const int  buf  = it & 1;
        const int  nbuf = buf ^ 1;
        const bool hasNext = (it + 1) < (K_LEN / 32);
        const int  kn = k0 + 32;

        // (1) Issue next A-tile global loads into registers (latency hidden
        //     behind the WMMA burst below) + prefetch one tile further out.
        float fa[16];
        if (hasNext) {
#pragma unroll
            for (int i = 0; i < 16; ++i) {
                const int e = tid + i * 256;
                fa[i] = tmpl[tBase + (size_t)(e >> 5) * rStride + kn + (e & 31)];
            }
            if (kn + 32 < K_LEN)
                __builtin_prefetch(&tmpl[tBase + (size_t)(tid >> 5) * rStride + kn + 32],
                                   0, 1);   // -> global_prefetch_b8
        }

        // (2) Fragment loads + WMMA on the current buffer.
        // B fragments (32x16, K x N): lane n = lane&15; same k order as A (h0=k[8h..], h1=k[16+8h..]).
        Frag bh[4], bl[4];
#pragma unroll
        for (int ni = 0; ni < 4; ++ni) {
            const int col = wn * 64 + ni * 16 + laneN;
            const __bf16* ph = &sBHi[buf][col * 32 + laneHalf * 8];
            const __bf16* pl = &sBLo[buf][col * 32 + laneHalf * 8];
            bh[ni].h[0] = *(const bf16x8*)(ph);
            bh[ni].h[1] = *(const bf16x8*)(ph + 16);
            bl[ni].h[0] = *(const bf16x8*)(pl);
            bl[ni].h[1] = *(const bf16x8*)(pl + 16);
        }
#pragma unroll
        for (int mi = 0; mi < 2; ++mi) {
            // A fragment (16x32, M x K): lane M = lane&15; ISA layout:
            // lanes 0-15: h0=k[0..7],  h1=k[16..23]
            // lanes16-31: h0=k[8..15], h1=k[24..31]
            const int row = wm * 32 + mi * 16 + laneN;
            Frag ah, al;
            const __bf16* ph = &sAHi[buf][row * 32 + laneHalf * 8];
            const __bf16* pl = &sALo[buf][row * 32 + laneHalf * 8];
            ah.h[0] = *(const bf16x8*)(ph);
            ah.h[1] = *(const bf16x8*)(ph + 16);
            al.h[0] = *(const bf16x8*)(pl);
            al.h[1] = *(const bf16x8*)(pl + 16);
#pragma unroll
            for (int ni = 0; ni < 4; ++ni) {
                acc[mi][ni] = __builtin_amdgcn_wmma_f32_16x16x32_bf16(
                    false, ah.v, false, bh[ni].v, (short)0, acc[mi][ni], false, false);
                acc[mi][ni] = __builtin_amdgcn_wmma_f32_16x16x32_bf16(
                    false, ah.v, false, bl[ni].v, (short)0, acc[mi][ni], false, false);
                acc[mi][ni] = __builtin_amdgcn_wmma_f32_16x16x32_bf16(
                    false, al.v, false, bh[ni].v, (short)0, acc[mi][ni], false, false);
            }
        }

        // (3) Convert + pack next tile into the other buffer.
        if (hasNext) {
#pragma unroll
            for (int i = 0; i < 16; ++i) {
                const int e = tid + i * 256;
                const int r = e >> 5;
                const int k = e & 31;
                const __bf16 ahi = (__bf16)fa[i];
                sAHi[nbuf][e] = ahi;
                sALo[nbuf][e] = (__bf16)(fa[i] - (float)ahi);
                sBHi[nbuf][e] = sGHi[kn + r + k];
                sBLo[nbuf][e] = sGLo[kn + r + k];
            }
        }
        // Single barrier: publishes next buffer AND protects current buffer
        // from being overwritten in the next iteration.
        __syncthreads();
    }

    // Epilogue: scale by 1/hh_sqrt, non-temporal f32 stores (pure streaming
    // output; keep L2 for the template). C/D layout: VGPR r -> M = r + 8*laneHalf.
#pragma unroll
    for (int mi = 0; mi < 2; ++mi)
#pragma unroll
        for (int ni = 0; ni < 4; ++ni) {
            const size_t j = (size_t)j0 + wn * 64 + ni * 16 + laneN;
#pragma unroll
            for (int r = 0; r < 8; ++r) {
                const int row = wm * 32 + mi * 16 + laneHalf * 8 + r;
                const size_t t = (size_t)t0 + row;
                __builtin_nontemporal_store(
                    acc[mi][ni][r] * sRcp[row],
                    &out[(((size_t)b * T_LEN + t) * 2 + c) * D_LEN + j]);
            }
        }
}

// ---------------------------------------------------------------------------
extern "C" void kernel_launch(void* const* d_in, const int* in_sizes, int n_in,
                              void* d_out, int out_size, void* d_ws, size_t ws_size,
                              hipStream_t stream) {
    (void)in_sizes; (void)n_in; (void)out_size; (void)ws_size;
    const float* X    = (const float*)d_in[0];  // (8, 2, 32768)
    const float* tmpl = (const float*)d_in[1];  // (4096, 2, 4096)
    const float* hh   = (const float*)d_in[2];  // (4096, 2)
    const float* S    = (const float*)d_in[3];  // (1, 1, 32768)
    float* out = (float*)d_out;                 // (8, 4096, 2, 32768)

    __bf16* gHi = (__bf16*)d_ws;                         // 16 * 36864 bf16
    __bf16* gLo = gHi + (size_t)16 * G_LEN;              // total ws: ~2.3 MB

    mf_stage1<<<dim3(D_LEN / 256, 16), 256, 0, stream>>>(X, S, gHi, gLo);
    mf_stage2<<<dim3(D_LEN / 128, T_LEN / 128, 16), 256, 0, stream>>>(
        tmpl, hh, gHi, gLo, out);
}
